// BaselineApproxLinearFP16_43602507989108
// MI455X (gfx1250) — matrix operand order repaired and checked by reading, not compiled, not run on hardware
//
#include <hip/hip_runtime.h>

typedef _Float16 half_t;
typedef __attribute__((ext_vector_type(8)))  _Float16 v8h;
typedef __attribute__((ext_vector_type(16))) _Float16 v16h;
typedef __attribute__((ext_vector_type(8)))  float    v8f;

// Problem size (fixed by the reference)
#define M_DIM 8192   // B * S
#define N_DIM 4096   // D_OUT
#define K_DIM 4096   // D_IN

// Tiling
constexpr int BM  = 128;          // block tile M
constexpr int BN  = 128;          // block tile N
constexpr int BK  = 64;           // block tile K (2 wmma k-steps of 32)
constexpr int LDT = BK + 8;       // padded LDS row stride in halves (144B: 16B-aligned, conflict-free)

// ---------------------------------------------------------------------------
// CDNA5 async global->LDS copy (ASYNCcnt-tracked), CDNA5 ISA.
// Inline asm: portable across ROCm-7.2 / amdgpu-toolchain builtin arity skew.
// ---------------------------------------------------------------------------
__device__ __forceinline__ void async_load_b128(unsigned lds_off, const void* gaddr) {
    asm volatile("global_load_async_to_lds_b128 %0, %1, off"
                 :
                 : "v"(lds_off), "v"(gaddr)
                 : "memory");
}

__device__ __forceinline__ void wait_async_zero() {
    asm volatile("s_wait_asynccnt 0x0" ::: "memory");
}

// ---------------------------------------------------------------------------
// LDS fragment loaders following the CDNA5 wave32 WMMA VGPR layouts
// (CDNA5 ISA §7.12.2)
// ---------------------------------------------------------------------------

// A-matrix 16x32 f16 fragment. Lanes 0-15 hold M=lane, K = [0..7] and [16..23];
// lanes 16-31 hold M=lane-16, K = [8..15] and [24..31].
__device__ __forceinline__ v16h load_a_frag(const half_t* tile, int row, int kbase) {
    const v8h lo = *reinterpret_cast<const v8h*>(tile + row * LDT + kbase);
    const v8h hi = *reinterpret_cast<const v8h*>(tile + row * LDT + kbase + 16);
    v16h f;
#pragma unroll
    for (int i = 0; i < 8; ++i) { f[i] = lo[i]; f[i + 8] = hi[i]; }
    return f;
}

// B-matrix 32x16 f16 fragment. Lane holds column N=lane%16, with K=[0..7],[16..23]
// (lanes 0-15) or K=[8..15],[24..31] (lanes 16-31), matching the A fragment.
__device__ __forceinline__ v16h load_b_frag(const half_t* tile, int row, int kbase) {
    const v8h lo = *reinterpret_cast<const v8h*>(tile + row * LDT + kbase);
    const v8h hi = *reinterpret_cast<const v8h*>(tile + row * LDT + kbase + 16);
    v16h f;
#pragma unroll
    for (int i = 0; i < 8; ++i) { f[i] = lo[i]; f[i + 8] = hi[i]; }
    return f;
}

// ---------------------------------------------------------------------------
// Tiled fp16 GEMM + bias:  out[m, n] = sum_k X[m, k] * W[n, k] + bias[n]
// 256 threads = 8 wave32 waves; block tile 128x128, wave tile 32x64.
// Async double-buffered LDS pipeline: one s_wait_asynccnt + one barrier / K-tile.
// ---------------------------------------------------------------------------
__global__ __launch_bounds__(256)
void linear_fp16_wmma_kernel(const half_t* __restrict__ X,
                             const half_t* __restrict__ W,
                             const half_t* __restrict__ bias,
                             half_t* __restrict__ out) {
    __shared__ half_t As[2][BM * LDT];   // 2 x 18 KiB
    __shared__ half_t Bs[2][BN * LDT];   // 2 x 18 KiB   (total 72 KiB < 320 KiB/WGP)

    const int tid    = threadIdx.x;
    const int lane   = tid & 31;
    const int lane16 = lane & 15;
    const int laneHi = lane >> 4;         // 0: lanes 0-15, 1: lanes 16-31
    const int wave   = tid >> 5;          // 0..7
    const int wRow   = (wave >> 1) * 32;  // wave tile row inside block (4 waves in M)
    const int wCol   = (wave & 1) * 64;   // wave tile col inside block (2 waves in N)

    const int mBase = blockIdx.y * BM;    // global row of block tile
    const int nBase = blockIdx.x * BN;    // global col of block tile

    // Global->LDS staging pattern: thread t moves 4 x 16B per tile per matrix.
    const int ldRow = tid >> 3;           // 0..31 (row within a 32-row pass)
    const int ldCol = (tid & 7) * 8;      // k offset in halves, 16B granular

    const half_t* aBase = X + (size_t)(mBase + ldRow) * K_DIM + ldCol;
    const half_t* bBase = W + (size_t)(nBase + ldRow) * K_DIM + ldCol;

    // 32-bit LDS byte offsets of this thread's store slots in each buffer.
    unsigned aLds[2], bLds[2];
#pragma unroll
    for (int b = 0; b < 2; ++b) {
        aLds[b] = (unsigned)(size_t)&As[b][ldRow * LDT + ldCol];
        bLds[b] = (unsigned)(size_t)&Bs[b][ldRow * LDT + ldCol];
    }

    // Issue one K-tile's worth of async copies (8 x b128 per thread).
    auto issue_tile = [&](int k0, int b) {
#pragma unroll
        for (int p = 0; p < 4; ++p) {
            async_load_b128(aLds[b] + (unsigned)(p * 32 * LDT * 2),
                            aBase + (size_t)(p * 32) * K_DIM + k0);
            async_load_b128(bLds[b] + (unsigned)(p * 32 * LDT * 2),
                            bBase + (size_t)(p * 32) * K_DIM + k0);
        }
    };

    v8f acc[2][4];
#pragma unroll
    for (int tm = 0; tm < 2; ++tm)
#pragma unroll
        for (int tn = 0; tn < 4; ++tn)
#pragma unroll
            for (int i = 0; i < 8; ++i) acc[tm][tn][i] = 0.0f;

    // Prologue: fill buffer 0.
    issue_tile(0, 0);
    wait_async_zero();
    __syncthreads();

    int buf = 0;
    for (int k0 = BK; k0 <= K_DIM; k0 += BK) {
        const bool more = (k0 < K_DIM);
        if (more) issue_tile(k0, buf ^ 1);   // prefetch next tile into other buffer

        const half_t* Atile = &As[buf][0];
        const half_t* Btile = &Bs[buf][0];

        // 2 k-steps of 32 -> 16 WMMAs per iteration per wave.
#pragma unroll
        for (int ks = 0; ks < BK; ks += 32) {
            v16h af[2], bf[4];
#pragma unroll
            for (int tm = 0; tm < 2; ++tm)
                af[tm] = load_a_frag(Atile, wRow + tm * 16 + lane16, ks + laneHi * 8);
#pragma unroll
            for (int tn = 0; tn < 4; ++tn)
                bf[tn] = load_b_frag(Btile, wCol + tn * 16 + lane16, ks + laneHi * 8);
#pragma unroll
            for (int tm = 0; tm < 2; ++tm)
#pragma unroll
                for (int tn = 0; tn < 4; ++tn)
                    acc[tm][tn] = __builtin_amdgcn_wmma_f32_16x16x32_f16(
                        /*neg_a=*/false, af[tm], /*neg_b=*/false, bf[tn],
                        /*c_mod=*/(short)0, acc[tm][tn],
                        /*reuse_a=*/false, /*reuse_b=*/false);
        }

        if (more) {
            wait_async_zero();   // this wave's async copies into buf^1 landed
            __syncthreads();     // everyone's copies landed, everyone done reading buf
            buf ^= 1;
        }
    }

    // Epilogue: C/D f32 16x16 layout -> lane = N column, VGPR r = M row
    // (lanes 16-31 carry rows M+8). Add bias, convert to fp16, store.
#pragma unroll
    for (int tm = 0; tm < 2; ++tm) {
#pragma unroll
        for (int tn = 0; tn < 4; ++tn) {
            const int n  = nBase + wCol + tn * 16 + lane16;
            const int m0 = mBase + wRow + tm * 16 + laneHi * 8;
            const float bf = (float)bias[n];
#pragma unroll
            for (int r = 0; r < 8; ++r) {
                out[(size_t)(m0 + r) * N_DIM + n] = (half_t)(acc[tm][tn][r] + bf);
            }
        }
    }
}

extern "C" void kernel_launch(void* const* d_in, const int* in_sizes, int n_in,
                              void* d_out, int out_size, void* d_ws, size_t ws_size,
                              hipStream_t stream) {
    (void)in_sizes; (void)n_in; (void)out_size; (void)d_ws; (void)ws_size;
    const half_t* X    = (const half_t*)d_in[0];  // (4, 2048, 4096) fp16
    const half_t* Wt   = (const half_t*)d_in[1];  // (4096, 4096) fp16, row-major (O, K)
    const half_t* bias = (const half_t*)d_in[2];  // (1, 4096) fp16
    half_t* out        = (half_t*)d_out;          // (4, 2048, 4096) fp16

    dim3 grid(N_DIM / BN, M_DIM / BM);  // (32, 64)
    dim3 block(256);
    linear_fp16_wmma_kernel<<<grid, block, 0, stream>>>(X, Wt, bias, out);
}
